// NaryTreeLSTM_19739669692603
// MI455X (gfx1250) — matrix operand order repaired and checked by reading, not compiled, not run on hardware
//
#include <hip/hip_runtime.h>
#include <hip/hip_bf16.h>
#include <math.h>

// ---------------- problem constants ----------------
#define L_NODES 8192
#define I_DIM   512
#define H_DIM   512
#define N_CH    4
#define PN      2048   // packed projection outputs: [xi|xf|xo|xu], each H
#define NH      2048   // N*H (flattened child state)
#define RN      3584   // recurrent gate rows: Ui(512)+Uo(512)+Uu(512)+Uf(2048)
#define MAXW    192    // max tree depth we schedule
#define AST     72     // LDS A-tile row stride (bf16 elems), padded: 36*m mod 64 all-distinct

typedef __bf16 bf16;
typedef __bf16 v16bf  __attribute__((ext_vector_type(16)));
typedef __bf16 bf16x8 __attribute__((ext_vector_type(8)));
typedef float  v8f    __attribute__((ext_vector_type(8)));
typedef int    v4i    __attribute__((ext_vector_type(4)));
typedef __attribute__((address_space(1))) v4i v4i_g;   // global (AS1)
typedef __attribute__((address_space(3))) v4i v4i_l;   // LDS (AS3)

#if __has_builtin(__builtin_amdgcn_global_load_async_to_lds_b128) && \
    __has_builtin(__builtin_amdgcn_s_wait_asynccnt)
#define ASYNC_LDS 1
#else
#define ASYNC_LDS 0
#endif

__device__ __forceinline__ float sigf(float x) { return 1.0f / (1.0f + __expf(-x)); }

__device__ __forceinline__ void async_wait0() {
#if ASYNC_LDS
    __builtin_amdgcn_s_wait_asynccnt(0);
#endif
}

// ---------------- conversion / packing ----------------
__global__ void cvt_bf16_kernel(bf16* __restrict__ dst, const float* __restrict__ src, int n) {
    int i = blockIdx.x * blockDim.x + threadIdx.x;
    int stride = gridDim.x * blockDim.x;
    for (; i < n; i += stride) dst[i] = (bf16)src[i];
}

__global__ void copy_f32_kernel(float* __restrict__ dst, const float* __restrict__ src, int n) {
    int i = blockIdx.x * blockDim.x + threadIdx.x;
    int stride = gridDim.x * blockDim.x;
    for (; i < n; i += stride) dst[i] = src[i];
}

// ---------------- WMMA tile helpers ----------------
// C[m][n] = sum_k A[m*K+k] * Bt[n*K+k]   (Bt stored N-major = "column major" B)
__device__ __forceinline__ v16bf make_a(bf16x8 lo, bf16x8 hi) {
    v16bf a;
#pragma unroll
    for (int i = 0; i < 8; ++i) { a[i] = lo[i]; a[8 + i] = hi[i]; }
    return a;
}

__device__ __forceinline__ void wmma_strip(const bf16* __restrict__ A,
                                           const bf16* __restrict__ Bt,
                                           int K, int row0, int col0, v8f& acc) {
    const int lane = threadIdx.x & 31;
    const int mn   = lane & 15;
    const int kh   = lane >> 4;
    const bf16* arow = A  + (size_t)(row0 + mn) * K;
    const bf16* brow = Bt + (size_t)(col0 + mn) * K;
#pragma unroll 2
    for (int k0 = 0; k0 < K; k0 += 32) {
        bf16x8 alo = *(const bf16x8*)(arow + k0 + kh * 16);
        bf16x8 ahi = *(const bf16x8*)(arow + k0 + kh * 16 + 8);
        v16bf  b   = *(const v16bf*)(brow + k0 + kh * 16);
        __builtin_prefetch(brow + k0 + 256, 0, 1);  // global_prefetch_b8
        v16bf a = make_a(alo, ahi);
        acc = __builtin_amdgcn_wmma_f32_16x16x32_bf16(false, a, false, b, (short)0, acc,
                                                      false, false);
    }
}

// D layout: lane holds N=lane&15; VGPR v holds M = v + 8*(lane>>4)
__device__ __forceinline__ void store_tile(float* __restrict__ C, int ldc,
                                           int row0, int col0, const v8f& acc,
                                           const float* __restrict__ bias) {
    const int lane = threadIdx.x & 31;
    const int n    = lane & 15;
    const int kh   = lane >> 4;
    const float bv = bias ? bias[col0 + n] : 0.0f;
#pragma unroll
    for (int v = 0; v < 8; ++v)
        C[(size_t)(row0 + v + 8 * kh) * ldc + col0 + n] = acc[v] + bv;
}

// ---------------- projection GEMM: xproj = nodes_bf16 @ Wpack^T + bias ----------------
__global__ __launch_bounds__(256)
void proj_gemm_kernel(const bf16* __restrict__ A, const bf16* __restrict__ Bt,
                      const float* __restrict__ bias, float* __restrict__ C) {
    const int wave = threadIdx.x >> 5;
    const int row0 = blockIdx.x * 16;
    const int col0 = blockIdx.y * 128 + wave * 16;
    v8f acc = {0.f, 0.f, 0.f, 0.f, 0.f, 0.f, 0.f, 0.f};
    wmma_strip(A, Bt, I_DIM, row0, col0, acc);
    store_tile(C, PN, row0, col0, acc, bias);
}

// ---------------- gate GEMM: PRE = [Hg|Cg] @ Upack^T ----------------
// Block tile 32M x 256N (8 waves = 2M x 4N, each wave 16M x 64N = 4 accumulators).
// A chunk (32 x 64 bf16) double-buffered in LDS, staged with async-load-to-LDS.
__device__ __forceinline__ void stage_chunkA(const bf16* __restrict__ Asrc, int row0, int kc,
                                             bf16* __restrict__ dstLds) {
    const int t  = threadIdx.x;
    const int m  = t >> 3;            // 0..31
    const int ks = (t & 7) * 8;       // 0..56
    const bf16* g = Asrc + (size_t)(row0 + m) * NH + kc + ks;
    bf16* l = dstLds + m * AST + ks;
#if ASYNC_LDS
    __builtin_amdgcn_global_load_async_to_lds_b128(
        (v4i_g*)(uintptr_t)g,
        (v4i_l*)(uint32_t)(uintptr_t)l,
        0, 0);
#else
    *(bf16x8*)l = *(const bf16x8*)g;
#endif
}

__global__ __launch_bounds__(256)
void gate_gemm_kernel(int w, const bf16* __restrict__ Hg, const bf16* __restrict__ Cg,
                      const bf16* __restrict__ U, float* __restrict__ PRE,
                      const int* __restrict__ cnt) {
    __shared__ bf16 Atile[2][32 * AST];
    const int B = cnt[w];
    const int wave = threadIdx.x >> 5;
    const int lane = threadIdx.x & 31;
    const int mw = wave >> 2;                 // 0..1 : M subtile
    const int nw = wave & 3;                  // 0..3 : N subtile
    const int n  = lane & 15;
    const int kh = lane >> 4;
    const int col0 = blockIdx.y * 256 + nw * 64;
    const bf16* A = (blockIdx.y < 6) ? Hg : Cg;   // i/o/u rows use h, f rows use c
    const bf16* br0 = U + (size_t)(col0 + 0  + n) * NH;
    const bf16* br1 = U + (size_t)(col0 + 16 + n) * NH;
    const bf16* br2 = U + (size_t)(col0 + 32 + n) * NH;
    const bf16* br3 = U + (size_t)(col0 + 48 + n) * NH;

    for (int mb = blockIdx.x; mb * 32 < B; mb += gridDim.x) {
        const int row0 = mb * 32;
        v8f acc0 = {0.f,0.f,0.f,0.f,0.f,0.f,0.f,0.f};
        v8f acc1 = acc0, acc2 = acc0, acc3 = acc0;
        stage_chunkA(A, row0, 0, Atile[0]);
        int buf = 0;
        for (int kc = 0; kc < NH; kc += 64) {
            async_wait0();
            __syncthreads();                        // chunk `buf` ready for all waves
            if (kc + 64 < NH) stage_chunkA(A, row0, kc + 64, Atile[buf ^ 1]);
            const bf16* lbase = &Atile[buf][(mw * 16 + (lane & 15)) * AST];
#pragma unroll
            for (int k0 = 0; k0 < 64; k0 += 32) {
                bf16x8 alo = *(const bf16x8*)(lbase + k0 + kh * 16);
                bf16x8 ahi = *(const bf16x8*)(lbase + k0 + kh * 16 + 8);
                v16bf a = make_a(alo, ahi);
                const int kk = kc + k0 + kh * 16;
                __builtin_prefetch(br0 + kk + 512, 0, 1);
                v16bf b0 = *(const v16bf*)(br0 + kk);
                v16bf b1 = *(const v16bf*)(br1 + kk);
                v16bf b2 = *(const v16bf*)(br2 + kk);
                v16bf b3 = *(const v16bf*)(br3 + kk);
                acc0 = __builtin_amdgcn_wmma_f32_16x16x32_bf16(false, a, false, b0, (short)0, acc0, false, false);
                acc1 = __builtin_amdgcn_wmma_f32_16x16x32_bf16(false, a, false, b1, (short)0, acc1, true,  false);
                acc2 = __builtin_amdgcn_wmma_f32_16x16x32_bf16(false, a, false, b2, (short)0, acc2, true,  false);
                acc3 = __builtin_amdgcn_wmma_f32_16x16x32_bf16(false, a, false, b3, (short)0, acc3, true,  false);
            }
            __syncthreads();                        // all waves done reading `buf`
            buf ^= 1;
        }
        const int row0w = row0 + mw * 16;
        store_tile(PRE, RN, row0w, col0 + 0,  acc0, nullptr);
        store_tile(PRE, RN, row0w, col0 + 16, acc1, nullptr);
        store_tile(PRE, RN, row0w, col0 + 32, acc2, nullptr);
        store_tile(PRE, RN, row0w, col0 + 48, acc3, nullptr);
    }
}

// ---------------- level scheduler: single workgroup, LDS-resident ----------------
__global__ __launch_bounds__(1024)
void schedule_kernel(const int* __restrict__ children,
                     int* __restrict__ bucket,
                     int* __restrict__ lvl_count, int* __restrict__ lvl_off) {
    __shared__ int lvl[L_NODES];       // 32 KB
    __shared__ int cnt[MAXW];
    __shared__ int off[MAXW + 1];
    const int tid = threadIdx.x;
    int ch[8][4];
#pragma unroll
    for (int s = 0; s < 8; ++s) {
        int i = tid + s * 1024;
        lvl[i] = 0;
#pragma unroll
        for (int k = 0; k < 4; ++k) ch[s][k] = children[i * 4 + k];
    }
    __syncthreads();
    for (int it = 0; it < MAXW; ++it) {   // monotone Jacobi: lvl = 1 + max(child lvls)
#pragma unroll
        for (int s = 0; s < 8; ++s) {
            int i = tid + s * 1024;
            int m = -1;
#pragma unroll
            for (int k = 0; k < 4; ++k) {
                int c = ch[s][k];
                if (c >= 0) { int lv = lvl[c]; m = lv > m ? lv : m; }
            }
            int nl = m + 1;
            if (nl > lvl[i]) lvl[i] = nl;
        }
        __syncthreads();
    }
    if (tid < MAXW) cnt[tid] = 0;
    __syncthreads();
#pragma unroll
    for (int s = 0; s < 8; ++s) {
        int i = tid + s * 1024;
        int lv = lvl[i]; if (lv > MAXW - 1) lv = MAXW - 1;
        lvl[i] = lv;
        atomicAdd(&cnt[lv], 1);
    }
    __syncthreads();
    if (tid == 0) {
        int acc = 0;
        for (int w = 0; w < MAXW; ++w) { off[w] = acc; acc += cnt[w]; }
        off[MAXW] = acc;
    }
    __syncthreads();
    if (tid < MAXW) { lvl_count[tid] = cnt[tid]; lvl_off[tid] = off[tid]; cnt[tid] = 0; }
    __syncthreads();
#pragma unroll
    for (int s = 0; s < 8; ++s) {
        int i = tid + s * 1024;
        int lv = lvl[i];
        int pos = atomicAdd(&cnt[lv], 1);
        bucket[off[lv] + pos] = i;
    }
}

// ---------------- per-wave: gather child states -> bf16 matrices ----------------
__global__ __launch_bounds__(256)
void gather_kernel(int w, const int* __restrict__ children,
                   const float* __restrict__ h_all, const float* __restrict__ c_all,
                   bf16* __restrict__ Hg, bf16* __restrict__ Cg,
                   const int* __restrict__ bucket,
                   const int* __restrict__ cnt, const int* __restrict__ off) {
    const int B = cnt[w];
    const int base = off[w];
    for (int b = blockIdx.x; b < B; b += gridDim.x) {
        const int idx = bucket[base + b];
        const int* chp = children + idx * 4;
        for (int e = threadIdx.x; e < NH; e += blockDim.x) {
            int k = e >> 9, j = e & 511;
            int c = chp[k];
            float hv = 0.f, cv = 0.f;
            if (c >= 0) {
                hv = h_all[(size_t)c * H_DIM + j];
                cv = c_all[(size_t)c * H_DIM + j];
            }
            Hg[(size_t)b * NH + e] = (bf16)hv;
            Cg[(size_t)b * NH + e] = (bf16)cv;
        }
    }
}

// ---------------- per-wave: elementwise LSTM cell update ----------------
__global__ __launch_bounds__(256)
void update_kernel(int w, const int* __restrict__ children,
                   const float* __restrict__ xproj, const float* __restrict__ PRE,
                   const int* __restrict__ bucket,
                   const int* __restrict__ cnt, const int* __restrict__ off,
                   float* __restrict__ c_all, float* __restrict__ h_out) {
    const int B = cnt[w];
    const int base = off[w];
    for (int b = blockIdx.x; b < B; b += gridDim.x) {
        const int idx = bucket[base + b];
        const float* pre = PRE + (size_t)b * RN;
        const float* xp  = xproj + (size_t)idx * PN;
        const int* chp = children + idx * 4;
        for (int j = threadIdx.x; j < H_DIM; j += blockDim.x) {
            float i_g = sigf(xp[j]        + pre[j]);          // Ui@h
            float o_g = sigf(xp[1024 + j] + pre[512 + j]);    // Uo@h
            float u_g = tanhf(xp[1536 + j] + pre[1024 + j]);  // Uu@h
            float xf  = xp[512 + j];
            float acc = i_g * u_g;
#pragma unroll
            for (int k = 0; k < 4; ++k) {
                int c = chp[k];
                float cc = (c >= 0) ? c_all[(size_t)c * H_DIM + j] : 0.f;
                float f_g = sigf(xf + pre[1536 + k * 512 + j]);  // Uf@c, row k*H+j
                acc += f_g * cc;
            }
            c_all[(size_t)idx * H_DIM + j] = acc;
            h_out[(size_t)idx * H_DIM + j] = o_g * tanhf(acc);
        }
    }
}

// ---------------- launcher ----------------
extern "C" void kernel_launch(void* const* d_in, const int* in_sizes, int n_in,
                              void* d_out, int out_size, void* d_ws, size_t ws_size,
                              hipStream_t stream) {
    const float* nodes    = (const float*)d_in[0];
    const int*   children = (const int*)d_in[1];
    const float* Wi = (const float*)d_in[2];  const float* bi = (const float*)d_in[3];
    const float* Wf = (const float*)d_in[4];  const float* bf = (const float*)d_in[5];
    const float* Wo = (const float*)d_in[6];  const float* bo = (const float*)d_in[7];
    const float* Wu = (const float*)d_in[8];  const float* bu = (const float*)d_in[9];
    const float* Ui = (const float*)d_in[10]; const float* Uf = (const float*)d_in[11];
    const float* Uu = (const float*)d_in[12]; const float* Uo = (const float*)d_in[13];
    float* h_out = (float*)d_out;

    char* ws = (char*)d_ws;
    size_t cur = 0;
    auto take = [&](size_t bytes) { size_t r = cur; cur = (cur + bytes + 255) & ~(size_t)255; return r; };
    bf16*  bnodes = (bf16*)(ws + take((size_t)L_NODES * I_DIM * sizeof(bf16)));
    bf16*  bW     = (bf16*)(ws + take((size_t)PN * I_DIM * sizeof(bf16)));
    float* biasP  = (float*)(ws + take((size_t)PN * sizeof(float)));
    bf16*  bU     = (bf16*)(ws + take((size_t)RN * NH * sizeof(bf16)));
    float* xproj  = (float*)(ws + take((size_t)L_NODES * PN * sizeof(float)));
    float* c_all  = (float*)(ws + take((size_t)L_NODES * H_DIM * sizeof(float)));
    bf16*  Hg     = (bf16*)(ws + take((size_t)L_NODES * NH * sizeof(bf16)));
    bf16*  Cg     = (bf16*)(ws + take((size_t)L_NODES * NH * sizeof(bf16)));
    float* PRE    = (float*)(ws + take((size_t)L_NODES * RN * sizeof(float)));
    int*   bucket    = (int*)(ws + take((size_t)L_NODES * sizeof(int)));
    int*   lvl_count = (int*)(ws + take((size_t)MAXW * sizeof(int)));
    int*   lvl_off   = (int*)(ws + take((size_t)MAXW * sizeof(int)));

    const int CB = 256;
    auto cgrid = [](size_t n) { size_t g = (n + 255) / 256; return (int)(g > 4096 ? 4096 : g); };

    size_t nNodes = (size_t)L_NODES * I_DIM;
    size_t nW = (size_t)H_DIM * I_DIM;
    cvt_bf16_kernel<<<cgrid(nNodes), CB, 0, stream>>>(bnodes, nodes, (int)nNodes);
    cvt_bf16_kernel<<<cgrid(nW), CB, 0, stream>>>(bW + 0 * nW, Wi, (int)nW);
    cvt_bf16_kernel<<<cgrid(nW), CB, 0, stream>>>(bW + 1 * nW, Wf, (int)nW);
    cvt_bf16_kernel<<<cgrid(nW), CB, 0, stream>>>(bW + 2 * nW, Wo, (int)nW);
    cvt_bf16_kernel<<<cgrid(nW), CB, 0, stream>>>(bW + 3 * nW, Wu, (int)nW);
    copy_f32_kernel<<<1, CB, 0, stream>>>(biasP + 0,    bi, H_DIM);
    copy_f32_kernel<<<1, CB, 0, stream>>>(biasP + 512,  bf, H_DIM);
    copy_f32_kernel<<<1, CB, 0, stream>>>(biasP + 1024, bo, H_DIM);
    copy_f32_kernel<<<1, CB, 0, stream>>>(biasP + 1536, bu, H_DIM);
    size_t nUsmall = (size_t)H_DIM * NH;
    size_t nUf = (size_t)NH * NH;
    cvt_bf16_kernel<<<cgrid(nUsmall), CB, 0, stream>>>(bU + 0 * nUsmall, Ui, (int)nUsmall);
    cvt_bf16_kernel<<<cgrid(nUsmall), CB, 0, stream>>>(bU + 1 * nUsmall, Uo, (int)nUsmall);
    cvt_bf16_kernel<<<cgrid(nUsmall), CB, 0, stream>>>(bU + 2 * nUsmall, Uu, (int)nUsmall);
    cvt_bf16_kernel<<<cgrid(nUf), CB, 0, stream>>>(bU + 3 * nUsmall, Uf, (int)nUf);

    proj_gemm_kernel<<<dim3(L_NODES / 16, PN / 128), 256, 0, stream>>>(bnodes, bW, biasP, xproj);

    schedule_kernel<<<1, 1024, 0, stream>>>(children, bucket, lvl_count, lvl_off);

    for (int w = 0; w < MAXW; ++w) {
        gather_kernel<<<512, 256, 0, stream>>>(w, children, h_out, c_all, Hg, Cg,
                                               bucket, lvl_count, lvl_off);
        gate_gemm_kernel<<<dim3(64, RN / 256), 256, 0, stream>>>(w, Hg, Cg, bU, PRE, lvl_count);
        update_kernel<<<512, 256, 0, stream>>>(w, children, xproj, PRE,
                                               bucket, lvl_count, lvl_off, c_all, h_out);
    }
}
